// Subgraphnet_83992380440733
// MI455X (gfx1250) — 8-bit matrix operand order repaired and checked by reading, compiled, not run on hardware
//
#include <hip/hip_runtime.h>
#include <hip/hip_bf16.h>
#include <math.h>
#include <stdint.h>

typedef __attribute__((ext_vector_type(8))) int   v8i;
typedef __attribute__((ext_vector_type(4))) int   v4i;
typedef __attribute__((ext_vector_type(8))) float v8f;
typedef __attribute__((ext_vector_type(2))) float v2f;

#define NN 4096
#define IN_DIM 128
#define DIM 256

#define GLOBAL_AS __attribute__((address_space(1)))
#define LDS_AS    __attribute__((address_space(3)))

__device__ __forceinline__ void async_b128(const void* gsrc, void* ldst) {
#if __has_builtin(__builtin_amdgcn_global_load_async_to_lds_b128)
    __builtin_amdgcn_global_load_async_to_lds_b128(
        (GLOBAL_AS v4i*)(uintptr_t)gsrc,
        (LDS_AS v4i*)(unsigned int)(uintptr_t)ldst, 0, 0);
#else
    *(float4*)ldst = *(const float4*)gsrc;
#endif
}

__device__ __forceinline__ void async_b32(const void* gsrc, void* ldst) {
#if __has_builtin(__builtin_amdgcn_global_load_async_to_lds_b32)
    __builtin_amdgcn_global_load_async_to_lds_b32(
        (GLOBAL_AS int*)(uintptr_t)gsrc,
        (LDS_AS int*)(unsigned int)(uintptr_t)ldst, 0, 0);
#else
    *(unsigned int*)ldst = *(const unsigned int*)gsrc;
#endif
}

__device__ __forceinline__ void async_wait0() {
#if __has_builtin(__builtin_amdgcn_s_wait_asynccnt)
    __builtin_amdgcn_s_wait_asynccnt(0);
#else
    asm volatile("s_wait_asynccnt 0" ::: "memory");
#endif
}

__global__ void binarize_kernel(const float* __restrict__ edge,
                                unsigned char* __restrict__ un,
                                unsigned char* __restrict__ unT) {
    __shared__ unsigned char tile[32][33];
    const int bx = blockIdx.x * 32;
    const int by = blockIdx.y * 32;
    const int tx = threadIdx.x, ty = threadIdx.y;
#pragma unroll
    for (int r = 0; r < 32; r += 8) {
        float v = edge[(size_t)(by + ty + r) * NN + bx + tx];
        unsigned char b = (v != 0.0f) ? 1 : 0;
        tile[ty + r][tx] = b;
        un[(size_t)(by + ty + r) * NN + bx + tx] = b;
    }
    __syncthreads();
#pragma unroll
    for (int r = 0; r < 32; r += 8) {
        unT[(size_t)(bx + ty + r) * NN + by + tx] = tile[tx][ty + r];
    }
}

__global__ void twohop_kernel(const unsigned char* __restrict__ un,
                              const unsigned char* __restrict__ unT,
                              unsigned char* __restrict__ two_hop) {
    const int wave = (int)((blockIdx.x * blockDim.x + threadIdx.x) >> 5);
    const int lane = threadIdx.x & 31;
    const int tiles = NN / 16;
    const int tm = wave / tiles, tn = wave % tiles;
    if (tm >= tiles) return;
    const int row  = tm * 16 + (lane & 15);
    const int col  = tn * 16 + (lane & 15);
    const int hiA  = (lane >> 4) ? 8 : 0;
    const int hiB  = (lane >> 4) ? 16 : 0;
    const unsigned char* arow = un  + (size_t)row * NN;
    const unsigned char* bcol = unT + (size_t)col * NN;
    v8i acc = {};
    for (int k0 = 0; k0 < NN; k0 += 64) {
        v8i a, b;
#pragma unroll
        for (int g = 0; g < 4; ++g) {
            uint2 v = *(const uint2*)(arow + k0 + 32 * (g >> 1) + 8 * (g & 1) + 2 * hiA);
            a[2 * g] = (int)v.x; a[2 * g + 1] = (int)v.y;
        }
#pragma unroll
        for (int g = 0; g < 2; ++g) {
            uint4 v = *(const uint4*)(bcol + k0 + 32 * g + hiB);
            b[4 * g] = (int)v.x; b[4 * g + 1] = (int)v.y;
            b[4 * g + 2] = (int)v.z; b[4 * g + 3] = (int)v.w;
        }
        acc = __builtin_amdgcn_wmma_i32_16x16x64_iu8(false, a, false, b, acc,
                                                     false, false);
    }
    const int mbase = tm * 16 + ((lane >> 4) ? 8 : 0);
#pragma unroll
    for (int r = 0; r < 8; ++r)
        two_hop[(size_t)(mbase + r) * NN + col] = (acc[r] != 0) ? 1 : 0;
}

__global__ void fc_kernel(const float* __restrict__ feat,
                          const float* __restrict__ fc_w,
                          const float* __restrict__ fc_b,
                          float* __restrict__ org_h) {
    const int wave = (int)((blockIdx.x * blockDim.x + threadIdx.x) >> 5);
    const int lane = threadIdx.x & 31;
    const int tilesN = DIM / 16;
    const int tm = wave / tilesN, tn = wave % tilesN;
    if (tm >= NN / 16) return;
    const int row  = tm * 16 + (lane & 15);
    const int colN = tn * 16 + (lane & 15);
    const int koff = (lane >> 4) ? 2 : 0;
    const float* arow = feat + (size_t)row * IN_DIM;
    v8f acc = {};
    for (int k0 = 0; k0 < IN_DIM; k0 += 4) {
        v2f a, b;
        a[0] = arow[k0 + koff];
        a[1] = arow[k0 + koff + 1];
        b[0] = fc_w[(size_t)(k0 + koff) * DIM + colN];
        b[1] = fc_w[(size_t)(k0 + koff + 1) * DIM + colN];
        acc = __builtin_amdgcn_wmma_f32_16x16x4_f32(false, a, false, b,
                                                    (short)0, acc, false, false);
    }
    const float bias  = fc_b[colN];
    const int   mbase = tm * 16 + ((lane >> 4) ? 8 : 0);
#pragma unroll
    for (int r = 0; r < 8; ++r) {
        float v = acc[r] + bias;
        org_h[(size_t)(mbase + r) * DIM + colN] = v > 0.0f ? v : 0.0f;
    }
}

__global__ void scores_kernel(const float* __restrict__ org_h,
                              const float* __restrict__ pw,
                              const float* __restrict__ pb,
                              unsigned long long* __restrict__ keys) {
    const int row  = (int)((blockIdx.x * blockDim.x + threadIdx.x) >> 5);
    const int lane = threadIdx.x & 31;
    if (row >= NN) return;
    const float* hr = org_h + (size_t)row * DIM;
    float s = 0.0f;
#pragma unroll
    for (int j = 0; j < 8; ++j) s += hr[lane + 32 * j] * pw[lane + 32 * j];
#pragma unroll
    for (int off = 16; off > 0; off >>= 1) s += __shfl_xor(s, off, 32);
    if (lane == 0) {
        float sc = 1.0f / (1.0f + __expf(-(s + pb[0])));
        unsigned int fb = __float_as_uint(sc);
        keys[row] = ((unsigned long long)fb << 32) |
                    (unsigned long long)(~(unsigned int)row);
    }
}

__global__ void __launch_bounds__(1024)
sort_kernel(const unsigned long long* __restrict__ keys,
            int* __restrict__ idx_out, float* __restrict__ val_out) {
    __shared__ unsigned long long sk[NN];
    const int tid = threadIdx.x;
    for (int i = tid; i < NN; i += 1024) sk[i] = keys[i];
    __syncthreads();
    for (int k = 2; k <= NN; k <<= 1) {
        for (int j = k >> 1; j > 0; j >>= 1) {
            for (int t = tid; t < NN; t += 1024) {
                int ixj = t ^ j;
                if (ixj > t) {
                    unsigned long long a = sk[t], b = sk[ixj];
                    bool desc = ((t & k) == 0);
                    if (desc ? (a < b) : (a > b)) { sk[t] = b; sk[ixj] = a; }
                }
            }
            __syncthreads();
        }
    }
    for (int i = tid; i < NN; i += 1024) {
        unsigned long long key = sk[i];
        idx_out[i] = (int)(~(unsigned int)key);
        val_out[i] = __uint_as_float((unsigned int)(key >> 32));
    }
}

__global__ void gather_h_kernel(const float* __restrict__ org_h,
                                const int* __restrict__ idx,
                                const float* __restrict__ vals,
                                float* __restrict__ h0, int kk) {
    const int r = blockIdx.y, c = threadIdx.x;
    float v = 0.0f;
    if (r < kk) v = org_h[(size_t)idx[r] * DIM + c] * vals[r];
    h0[(size_t)r * DIM + c] = v;
}

__global__ void gather_g_kernel(const unsigned char* __restrict__ two_hop,
                                const int* __restrict__ idx,
                                unsigned char* __restrict__ g_sel,
                                int kk, int kkp) {
    const int r = blockIdx.y;
    const int c = blockIdx.x * 256 + threadIdx.x;
    if (c >= kkp) return;
    unsigned char v = 0;
    if (r < kk && c < kk) v = two_hop[(size_t)idx[r] * NN + idx[c]];
    g_sel[(size_t)r * kkp + c] = v;
}

__global__ void __launch_bounds__(512)
prop_kernel(const unsigned char* __restrict__ g,
            const float* __restrict__ h,
            float* __restrict__ hi, int kkp) {
    __shared__ float         hbuf[16][DIM];
    __shared__ unsigned char gbuf[16][16];
    const int tm   = blockIdx.x;
    const int tid  = threadIdx.x;
    const int wave = tid >> 5;
    const int lane = tid & 31;
    const int colN = wave * 16 + (lane & 15);
    const int koff = (lane >> 4) ? 2 : 0;
    const unsigned char* gbase = g + (size_t)(tm * 16) * kkp;
    v8f acc = {};
    for (int k0 = 0; k0 < kkp; k0 += 16) {
        const float* hsrc = h + (size_t)k0 * DIM;
#pragma unroll
        for (int i = 0; i < 2; ++i) {
            int flat4 = tid + 512 * i;
            int row   = flat4 >> 6;
            int c4    = (flat4 & 63) << 2;
            async_b128(hsrc + (size_t)row * DIM + c4, &hbuf[row][c4]);
        }
        if (tid < 64) {
            int row = tid >> 2, b4 = (tid & 3) << 2;
            async_b32(gbase + (size_t)row * kkp + k0 + b4,
                      (unsigned int*)&gbuf[row][b4]);
        }
        async_wait0();
        __syncthreads();
#pragma unroll
        for (int ks = 0; ks < 16; ks += 4) {
            v2f a, b;
            a[0] = (float)gbuf[lane & 15][ks + koff];
            a[1] = (float)gbuf[lane & 15][ks + koff + 1];
            b[0] = hbuf[ks + koff][colN];
            b[1] = hbuf[ks + koff + 1][colN];
            acc = __builtin_amdgcn_wmma_f32_16x16x4_f32(false, a, false, b,
                                                        (short)0, acc, false,
                                                        false);
        }
        __syncthreads();
    }
    const int mbase = tm * 16 + ((lane >> 4) ? 8 : 0);
#pragma unroll
    for (int r = 0; r < 8; ++r)
        hi[(size_t)(mbase + r) * DIM + colN] = acc[r];
}

__global__ void gcnii_kernel(const float* __restrict__ hi,
                             const float* __restrict__ h0,
                             const float* __restrict__ h,
                             const float* __restrict__ W,
                             float* __restrict__ hnext,
                             int kkp, float theta) {
    const int wave = (int)((blockIdx.x * blockDim.x + threadIdx.x) >> 5);
    const int lane = threadIdx.x & 31;
    const int tilesN = DIM / 16;
    const int tm = wave / tilesN, tn = wave % tilesN;
    if (tm >= kkp / 16) return;
    const int row  = tm * 16 + (lane & 15);
    const int colN = tn * 16 + (lane & 15);
    const int koff = (lane >> 4) ? 2 : 0;
    const float* a_hi = hi + (size_t)row * DIM;
    const float* a_h0 = h0 + (size_t)row * DIM;
    v8f acc = {};
    for (int k0 = 0; k0 < DIM; k0 += 4) {
        v2f a, b;
        a[0] = a_hi[k0 + koff];
        a[1] = a_hi[k0 + koff + 1];
        b[0] = W[(size_t)(k0 + koff) * DIM + colN];
        b[1] = W[(size_t)(k0 + koff + 1) * DIM + colN];
        acc = __builtin_amdgcn_wmma_f32_16x16x4_f32(false, a, false, b,
                                                    (short)0, acc, false, false);
    }
    for (int k0 = 0; k0 < DIM; k0 += 4) {
        v2f a, b;
        a[0] = a_h0[k0 + koff];
        a[1] = a_h0[k0 + koff + 1];
        b[0] = W[(size_t)(DIM + k0 + koff) * DIM + colN];
        b[1] = W[(size_t)(DIM + k0 + koff + 1) * DIM + colN];
        acc = __builtin_amdgcn_wmma_f32_16x16x4_f32(false, a, false, b,
                                                    (short)0, acc, false, false);
    }
    const int mbase = tm * 16 + ((lane >> 4) ? 8 : 0);
#pragma unroll
    for (int r = 0; r < 8; ++r) {
        size_t o = (size_t)(mbase + r) * DIM + colN;
        float vhi = hi[o], vh0 = h0[o], vh = h[o];
        float t = theta * acc[r] + (1.0f - theta) * (0.9f * vhi + 0.1f * vh0)
                  + 2.0f * vh;
        hnext[o] = t > 0.0f ? t : 0.0f;
    }
}

__global__ void zero_out_kernel(float* __restrict__ out) {
    int t = blockIdx.x * blockDim.x + threadIdx.x;
    if (t < NN * DIM) out[t] = 0.0f;
}

__global__ void scatter_max_kernel(const float* __restrict__ h,
                                   const int* __restrict__ idx,
                                   float* __restrict__ out, int kk) {
    const int r = blockIdx.y, c = threadIdx.x;
    if (r >= kk) return;
    size_t o = (size_t)idx[r] * DIM + c;
    float v = h[(size_t)r * DIM + c];
    float cur = out[o];
    out[o] = v > cur ? v : cur;
}

extern "C" void kernel_launch(void* const* d_in, const int* in_sizes, int n_in,
                              void* d_out, int out_size, void* d_ws, size_t ws_size,
                              hipStream_t stream) {
    const float* feat   = (const float*)d_in[0];
    const float* edge   = (const float*)d_in[1];
    const float* fc_w   = (const float*)d_in[3];
    const float* fc_b   = (const float*)d_in[4];
    const float* proj_w = (const float*)d_in[5];
    const float* proj_b = (const float*)d_in[6];
    const float* gcn_w  = (const float*)d_in[7];
    float* out = (float*)d_out;

    char* ws = (char*)d_ws;
    size_t off = 0;
    auto alloc = [&](size_t bytes) {
        size_t o = off;
        off += (bytes + 255) & ~(size_t)255;
        return o;
    };
    const int KKP_MAX = 3696;
    unsigned char* un      = (unsigned char*)(ws + alloc((size_t)NN * NN));
    unsigned char* unT     = (unsigned char*)(ws + alloc((size_t)NN * NN));
    unsigned char* two_hop = (unsigned char*)(ws + alloc((size_t)NN * NN));
    float* org_h  = (float*)(ws + alloc((size_t)NN * DIM * 4));
    unsigned long long* keys = (unsigned long long*)(ws + alloc((size_t)NN * 8));
    int*   idx    = (int*)  (ws + alloc((size_t)NN * 4));
    float* vals   = (float*)(ws + alloc((size_t)NN * 4));
    float* h0     = (float*)(ws + alloc((size_t)KKP_MAX * DIM * 4));
    float* hA     = (float*)(ws + alloc((size_t)KKP_MAX * DIM * 4));
    float* hB     = (float*)(ws + alloc((size_t)KKP_MAX * DIM * 4));
    float* hi     = (float*)(ws + alloc((size_t)KKP_MAX * DIM * 4));
    unsigned char* g_sel = (unsigned char*)(ws + alloc((size_t)KKP_MAX * KKP_MAX));
    (void)ws_size; (void)in_sizes; (void)n_in; (void)out_size;

    zero_out_kernel<<<(NN * DIM + 255) / 256, 256, 0, stream>>>(out);
    binarize_kernel<<<dim3(NN / 32, NN / 32), dim3(32, 8), 0, stream>>>(edge, un, unT);
    {
        int waves = (NN / 16) * (NN / 16);
        twohop_kernel<<<waves * 32 / 256, 256, 0, stream>>>(un, unT, two_hop);
    }
    {
        int waves = (NN / 16) * (DIM / 16);
        fc_kernel<<<waves * 32 / 256, 256, 0, stream>>>(feat, fc_w, fc_b, org_h);
    }

    const int   kks[3]    = {3686, 2867, 2457};
    const float thetas[3] = {logf(1.5f), logf(1.25f), logf(7.0f / 6.0f)};

    for (int lvl = 0; lvl < 3; ++lvl) {
        const int kk  = kks[lvl];
        const int kkp = (kk + 15) & ~15;

        scores_kernel<<<NN * 32 / 256, 256, 0, stream>>>(
            org_h, proj_w + (size_t)lvl * DIM, proj_b + lvl, keys);
        sort_kernel<<<1, 1024, 0, stream>>>(keys, idx, vals);
        gather_h_kernel<<<dim3(1, kkp), 256, 0, stream>>>(org_h, idx, vals, h0, kk);
        gather_g_kernel<<<dim3((kkp + 255) / 256, kkp), 256, 0, stream>>>(
            two_hop, idx, g_sel, kk, kkp);

        const float* hcur = h0;
        float* ping = hB;
        float* pong = hA;
        for (int j = 0; j < 3; ++j) {
            prop_kernel<<<kkp / 16, 512, 0, stream>>>(g_sel, hcur, hi, kkp);
            int waves  = (kkp / 16) * (DIM / 16);
            gcnii_kernel<<<waves * 32 / 256, 256, 0, stream>>>(
                hi, h0, hcur, gcn_w + (size_t)j * 2 * DIM * DIM, ping, kkp,
                thetas[j]);
            hcur = ping;
            float* t = ping; ping = pong; pong = t;
        }
        scatter_max_kernel<<<dim3(1, kk), 256, 0, stream>>>(hcur, idx, out, kk);
    }
}
